// DARTSCell_15187004358878
// MI455X (gfx1250) — compile-verified, matrix operand order checked by reading, not run on hardware
//
#include <hip/hip_runtime.h>

typedef __attribute__((ext_vector_type(16))) __bf16 v16bf;
typedef __attribute__((ext_vector_type(8)))  __bf16 v8bf;
typedef __attribute__((ext_vector_type(8)))  float  v8f;

#define WMMA_BF16(A, Bm, C) \
  __builtin_amdgcn_wmma_f32_16x16x32_bf16(false, (A), false, (Bm), (short)0, (C), false, false)

namespace {

constexpr int TT = 256;   // timesteps
constexpr int BB = 256;   // batch
constexpr int NH = 256;   // nhid
constexpr int NI = 256;   // ninp

constexpr int SB_COLS = 264;   // 256 + 8 pad; row = 528 B (16B aligned, banks spread)

constexpr int TILE_ELEMS = 512;                  // one 32Kx16N bf16 B-tile
constexpr int W0_ELEMS = 16 * 32 * TILE_ELEMS;   // 262144
constexpr int WS_ELEMS = 8  * 32 * TILE_ELEMS;   // 131072 per Ws matrix
constexpr int PACK_TOTAL = W0_ELEMS + 8 * WS_ELEMS;      // 1310720 bf16
constexpr size_t XBF_ELEMS = (size_t)TT * BB * NI;       // 16.7M bf16
constexpr size_t WS_NEEDED = ((size_t)PACK_TOTAL + XBF_ELEMS) * sizeof(__bf16);

// ---- fast gate math: v_tanh_f32 if available, else exp + fast rcp ----
#if __has_builtin(__builtin_amdgcn_tanhf)
__device__ __forceinline__ float tanh_fast(float x) {
  return __builtin_amdgcn_tanhf(x);
}
#else
__device__ __forceinline__ float tanh_fast(float x) {
  float a = fminf(fmaxf(-2.0f * x, -30.0f), 30.0f);
  float e = __expf(a);
  return (1.0f - e) * __builtin_amdgcn_rcpf(1.0f + e);
}
#endif
__device__ __forceinline__ float sigm(float x) {
  return fmaf(0.5f, tanh_fast(0.5f * x), 0.5f);   // 1 TRANS + 2 VALU
}
// ACT: 0=sigmoid 1=relu 2=tanh 3=identity
template <int ACT> __device__ __forceinline__ float act_fn(float x) {
  if constexpr (ACT == 0) return sigm(x);
  else if constexpr (ACT == 1) return fmaxf(x, 0.0f);
  else if constexpr (ACT == 2) return tanh_fast(x);
  else return x;
}

// GEMM-input states are 0,1,2,3,5. State 5 aliases slot 0 (lifetime-disjoint,
// fenced by the step-1/2/3 store barriers and the staging barrier).
__host__ __device__ constexpr int lds_slot(int s) { return s == 5 ? 0 : s; }

} // namespace

// ---------------------------------------------------------------------------
// One-shot weight repack: fp32 row-major -> bf16, pre-swizzled into WMMA
// B-fragment order (one contiguous 32B-per-lane load per fragment).
//   lanes 0-15 : N = lane,    elems 0-7 -> K 0-7,  elems 8-15 -> K 16-23
//   lanes 16-31: N = lane-16, elems 0-7 -> K 8-15, elems 8-15 -> K 24-31
// ---------------------------------------------------------------------------
__global__ __launch_bounds__(256)
void darts_pack_weights(const float* __restrict__ W0,
                        const float* __restrict__ Ws,
                        __bf16* __restrict__ out) {
  int gid = blockIdx.x * 256 + threadIdx.x;
  if (gid >= PACK_TOTAL) return;
  float v;
  if (gid < W0_ELEMS) {
    int rel = gid;
    int tile = rel >> 9;
    int in_tile = rel & 511;
    int lane = in_tile >> 4;
    int e = in_tile & 15;
    int kb = tile >> 5;
    int nt = tile & 31;
    int K = ((e < 8) ? e : 16 + (e - 8)) + ((lane >= 16) ? 8 : 0);
    int k = kb * 32 + K;
    int n = nt * 16 + (lane & 15);
    v = W0[(size_t)k * (2 * NH) + n];
  } else {
    int g = gid - W0_ELEMS;
    int i = g / WS_ELEMS;
    int rel = g - i * WS_ELEMS;
    int tile = rel >> 9;
    int in_tile = rel & 511;
    int lane = in_tile >> 4;
    int e = in_tile & 15;
    int kb = tile >> 5;
    int nt = tile & 31;
    int K = ((e < 8) ? e : 16 + (e - 8)) + ((lane >= 16) ? 8 : 0);
    int k = kb * 32 + K;
    int n = nt * 16 + (lane & 15);
    v = Ws[(size_t)i * NH * (2 * NH) + (size_t)k * (2 * NH) + n];
  }
  out[gid] = (__bf16)v;
}

// ---- one-shot x fp32 -> bf16 (feeds the async-to-LDS staging path) ----
__global__ __launch_bounds__(256)
void darts_convert_x(const float* __restrict__ x, __bf16* __restrict__ out) {
  size_t i = ((size_t)blockIdx.x * 256 + threadIdx.x) * 8;
  if (i >= XBF_ELEMS) return;
  const float4* p = (const float4*)(x + i);
  float4 a = p[0], b = p[1];
  __bf16* d = out + i;
  d[0] = (__bf16)a.x; d[1] = (__bf16)a.y; d[2] = (__bf16)a.z; d[3] = (__bf16)a.w;
  d[4] = (__bf16)b.x; d[5] = (__bf16)b.y; d[6] = (__bf16)b.z; d[7] = (__bf16)b.w;
}

// ---------------------------------------------------------------------------
// Shared 8-K-block GEMM piece: 4 independent WMMA accumulator chains.
// ---------------------------------------------------------------------------
__device__ __forceinline__ void gemm8(const __bf16 (*A)[SB_COLS],
                                      const __bf16* __restrict__ wpk,
                                      v8f& aC0, v8f& aC1, v8f& aH0, v8f& aH1,
                                      int m, int kofs, int w, int lane) {
#pragma unroll
  for (int kb2 = 0; kb2 < 4; ++kb2) {
    const int kb = kb2 * 2;
    union { v16bf v; v8bf h[2]; } a0, a1;
    const __bf16* ar0 = &A[m][kb * 32 + kofs];
    const __bf16* ar1 = &A[m][(kb + 1) * 32 + kofs];
    a0.h[0] = *(const v8bf*)(ar0);
    a0.h[1] = *(const v8bf*)(ar0 + 16);
    a1.h[0] = *(const v8bf*)(ar1);
    a1.h[1] = *(const v8bf*)(ar1 + 16);
    v16bf bc0 = *(const v16bf*)(wpk + (size_t)(kb * 32 + w) * TILE_ELEMS + lane * 16);
    v16bf bh0 = *(const v16bf*)(wpk + (size_t)(kb * 32 + 16 + w) * TILE_ELEMS + lane * 16);
    v16bf bc1 = *(const v16bf*)(wpk + (size_t)((kb + 1) * 32 + w) * TILE_ELEMS + lane * 16);
    v16bf bh1 = *(const v16bf*)(wpk + (size_t)((kb + 1) * 32 + 16 + w) * TILE_ELEMS + lane * 16);
    aC0 = WMMA_BF16(a0.v, bc0, aC0);
    aH0 = WMMA_BF16(a0.v, bh0, aH0);
    aC1 = WMMA_BF16(a1.v, bc1, aC1);
    aH1 = WMMA_BF16(a1.v, bh1, aH1);
  }
}

// ---------------------------------------------------------------------------
// One DARTS step; barrier only when the produced state is stored to LDS.
// ---------------------------------------------------------------------------
template <int PRED, int ACT, int CUR, bool STORE>
__device__ __forceinline__ void rnn_step(const __bf16* __restrict__ wpk,
                                         __bf16 (*sb)[16][SB_COLS],
                                         float (&st)[9][8],
                                         int m, int kofs, int w, int lane,
                                         int rbase, int col) {
  v8f aC0 = {}, aC1 = {}, aH0 = {}, aH1 = {};
  gemm8(sb[lds_slot(PRED)], wpk, aC0, aC1, aH0, aH1, m, kofs, w, lane);
  v8f accC = aC0 + aC1;
  v8f accH = aH0 + aH1;
#pragma unroll
  for (int r = 0; r < 8; ++r) {
    float c = sigm(accC[r]);
    float h = act_fn<ACT>(accH[r]);
    float sp = st[PRED][r];
    float s = fmaf(c, h - sp, sp);
    st[CUR][r] = s;
    if constexpr (STORE) sb[lds_slot(CUR)][rbase + r][col] = (__bf16)s;
  }
  if constexpr (STORE) __syncthreads();
}

// ---------------------------------------------------------------------------
// Persistent RNN: 16 blocks (one 16-row batch tile each) x 512 threads
// (16 wave32s; wave w owns output columns [16w,16w+16)).
// ASYNC=true: x tiles staged one timestep ahead via global_load_async_to_lds
// (ASYNCcnt — decoupled from the LOADcnt waits of the WMMA B-fragments).
// ---------------------------------------------------------------------------
template <bool ASYNC>
__global__ __launch_bounds__(512, 1)
void darts_rnn_persistent(const float* __restrict__ x,
                          const __bf16* __restrict__ xbf,
                          const float* __restrict__ h0g,
                          const __bf16* __restrict__ w0p,
                          const __bf16* __restrict__ wsp,
                          float* __restrict__ out) {
  __shared__ __bf16 xbuf[2][16][SB_COLS];   // double-buffered x tile (bf16)
  __shared__ __bf16 hbuf[16][SB_COLS];      // h_prev (bf16) for GEMM0
  __shared__ __bf16 sbuf[4][16][SB_COLS];   // states 0/5, 1, 2, 3

  const int tid = threadIdx.x;
  const int w = tid >> 5;
  const int lane = tid & 31;
  const int m = lane & 15;
  const int hi = lane >> 4;
  const int kofs = hi * 8;
  const int col = w * 16 + m;
  const int rbase = hi * 8;
  const int brow0 = blockIdx.x * 16;

  float hprev[8];
#pragma unroll
  for (int r = 0; r < 8; ++r) {
    hprev[r] = h0g[(size_t)(brow0 + rbase + r) * NH + col];
    hbuf[rbase + r][col] = (__bf16)hprev[r];
  }

  // x staging resources
  const int sx_idx = tid * 8;               // 512 thr x 8 = 16x256 (reg path)
  const int sx_r = sx_idx >> 8;
  const int sx_c = sx_idx & 255;
  const float* xbase = x + (size_t)(brow0 + sx_r) * NI + sx_c;
  float4 xa{}, xb{};

  // async path: wave w copies row w; lane covers 16B (8 bf16)
  auto issue_async_x = [&](int t) {
    const __bf16* g = xbf + ((size_t)t * BB + brow0 + w) * NI + lane * 8;
    unsigned ldsoff = (unsigned)(uintptr_t)&xbuf[t & 1][w][lane * 8];
    unsigned long long ga = (unsigned long long)(uintptr_t)g;
    asm volatile("global_load_async_to_lds_b128 %0, %1, off"
                 :: "v"(ldsoff), "v"(ga) : "memory");
  };

  if constexpr (ASYNC) {
    issue_async_x(0);
  } else {
    const float4* p = (const float4*)xbase;
    xa = p[0];
    xb = p[1];
  }

  for (int t = 0; t < TT; ++t) {
    // ---- make x[t] tile visible in xbuf[t&1] ----
    if constexpr (ASYNC) {
      asm volatile("s_wait_asynccnt 0x0" ::: "memory");
    } else {
      __bf16* dst = &xbuf[t & 1][sx_r][sx_c];
      dst[0] = (__bf16)xa.x; dst[1] = (__bf16)xa.y;
      dst[2] = (__bf16)xa.z; dst[3] = (__bf16)xa.w;
      dst[4] = (__bf16)xb.x; dst[5] = (__bf16)xb.y;
      dst[6] = (__bf16)xb.z; dst[7] = (__bf16)xb.w;
    }
    __syncthreads();

    // ---- GEMM0: [x | h] @ W0 (2 x 8 K-blocks, 4 accumulator chains) ----
    v8f aC0 = {}, aC1 = {}, aH0 = {}, aH1 = {};
    gemm8(xbuf[t & 1], w0p, aC0, aC1, aH0, aH1, m, kofs, w, lane);
    gemm8(hbuf, w0p + (size_t)(8 * 32) * TILE_ELEMS, aC0, aC1, aH0, aH1,
          m, kofs, w, lane);
    v8f accC = aC0 + aC1;
    v8f accH = aH0 + aH1;

    float st[9][8];
#pragma unroll
    for (int r = 0; r < 8; ++r) {
      float c0 = sigm(accC[r]);
      float h0v = tanh_fast(accH[r]);
      float s0 = fmaf(c0, h0v - hprev[r], hprev[r]);
      st[0][r] = s0;
      sbuf[0][rbase + r][col] = (__bf16)s0;
    }
    __syncthreads();

    // ---- DARTS_V1 genotype (pred, act, cur, store) ----
    rnn_step<0, 0, 1, true >(wsp + 0 * WS_ELEMS, sbuf, st, m, kofs, w, lane, rbase, col); // sigmoid,0
    rnn_step<1, 1, 2, true >(wsp + 1 * WS_ELEMS, sbuf, st, m, kofs, w, lane, rbase, col); // relu,1
    rnn_step<1, 1, 3, true >(wsp + 2 * WS_ELEMS, sbuf, st, m, kofs, w, lane, rbase, col); // relu,1
    rnn_step<1, 3, 4, false>(wsp + 3 * WS_ELEMS, sbuf, st, m, kofs, w, lane, rbase, col); // identity,1
    rnn_step<2, 2, 5, true >(wsp + 4 * WS_ELEMS, sbuf, st, m, kofs, w, lane, rbase, col); // tanh,2
    rnn_step<5, 0, 6, false>(wsp + 5 * WS_ELEMS, sbuf, st, m, kofs, w, lane, rbase, col); // sigmoid,5
    rnn_step<3, 2, 7, false>(wsp + 6 * WS_ELEMS, sbuf, st, m, kofs, w, lane, rbase, col); // tanh,3
    rnn_step<5, 1, 8, false>(wsp + 7 * WS_ELEMS, sbuf, st, m, kofs, w, lane, rbase, col); // relu,5

    // ---- issue x[t+1] staging now (younger than all GEMM waits of step t) ----
    if (t + 1 < TT) {
      if constexpr (ASYNC) {
        issue_async_x(t + 1);
      } else {
        const float4* p = (const float4*)(xbase + (size_t)(t + 1) * BB * NI);
        xa = p[0];
        xb = p[1];
      }
    }

    // ---- h_new = mean(states 1..8); write out; feed recurrence ----
#pragma unroll
    for (int r = 0; r < 8; ++r) {
      float hn = (st[1][r] + st[2][r] + st[3][r] + st[4][r] +
                  st[5][r] + st[6][r] + st[7][r] + st[8][r]) * 0.125f;
      size_t o = ((size_t)t * BB + brow0 + rbase + r) * NH + col;
      out[o] = hn;
      if (t == TT - 1)
        out[(size_t)TT * BB * NH + (size_t)(brow0 + rbase + r) * NH + col] = hn;
      hprev[r] = hn;
      hbuf[rbase + r][col] = (__bf16)hn;
    }
    // next conflicting LDS access is fenced by the staging barrier of t+1
  }
}

extern "C" void kernel_launch(void* const* d_in, const int* in_sizes, int n_in,
                              void* d_out, int out_size, void* d_ws, size_t ws_size,
                              hipStream_t stream) {
  (void)in_sizes; (void)n_in; (void)out_size;
  const float* x  = (const float*)d_in[0];   // [T,B,256] fp32
  const float* h0 = (const float*)d_in[1];   // [1,B,256] fp32
  const float* W0 = (const float*)d_in[2];   // [512,512] fp32
  const float* Ws = (const float*)d_in[3];   // [8,256,512] fp32
  // d_in[4] = rnn_mask (unused)

  __bf16* wpack = (__bf16*)d_ws;             // 2.5 MB packed bf16 weights
  darts_pack_weights<<<(PACK_TOTAL + 255) / 256, 256, 0, stream>>>(W0, Ws, wpack);

  if (ws_size >= WS_NEEDED) {
    __bf16* xbf = wpack + PACK_TOTAL;        // 33.5 MB bf16 copy of x
    darts_convert_x<<<(int)((XBF_ELEMS / 8 + 255) / 256), 256, 0, stream>>>(x, xbf);
    darts_rnn_persistent<true><<<16, 512, 0, stream>>>(x, xbf, h0, wpack,
                                                       wpack + W0_ELEMS,
                                                       (float*)d_out);
  } else {
    darts_rnn_persistent<false><<<16, 512, 0, stream>>>(x, nullptr, h0, wpack,
                                                        wpack + W0_ELEMS,
                                                        (float*)d_out);
  }
}
